// HEM__6390911336548
// MI455X (gfx1250) — compile-verified, matrix operand order checked by reading, not run on hardware
//
#include <hip/hip_runtime.h>
#include <hip/hip_bf16.h>
#include <stdint.h>

// Problem constants (shape (8, 3, 1024, 1024) fp32)
#define BATCH     8
#define CHAN      3
#define NPIX      (1024 * 1024)          // pixels per batch image
#define N4        (NPIX / 4)             // float4 elements per channel plane
#define HARD_IND  (NPIX / 2)             // int(0.5 * n)
#define NBINS     4096
#define BIN_SCALE 256.0f                 // bins cover [0, 16): r * 256 -> bin
#define RAND_CUT  429496729u             // 0.1 * 2^32  (p = 0.1 Bernoulli)

typedef __attribute__((ext_vector_type(16))) _Float16 v16h;
typedef __attribute__((ext_vector_type(8)))  float    v8f;

__device__ __forceinline__ uint32_t hash_u32(uint32_t s) {
    // Deterministic per-pixel PRNG (stand-in for jax.random.permutation(key=42))
    s ^= 42u * 0x9E3779B9u;
    s ^= s >> 16; s *= 0x7feb352du;
    s ^= s >> 15; s *= 0x846ca68bu;
    s ^= s >> 16;
    return s;
}

// channel-summed |x-y| for 4 consecutive pixels, from float4-vectorized planes
__device__ __forceinline__ void residual4(const float4& x0, const float4& x1,
                                          const float4& x2, const float4& y0,
                                          const float4& y1, const float4& y2,
                                          float r[4]) {
    r[0] = fabsf(x0.x - y0.x) + fabsf(x1.x - y1.x) + fabsf(x2.x - y2.x);
    r[1] = fabsf(x0.y - y0.y) + fabsf(x1.y - y1.y) + fabsf(x2.y - y2.y);
    r[2] = fabsf(x0.z - y0.z) + fabsf(x1.z - y1.z) + fabsf(x2.z - y2.z);
    r[3] = fabsf(x0.w - y0.w) + fabsf(x1.w - y1.w) + fabsf(x2.w - y2.w);
}

// ---------------------------------------------------------------------------
// Kernel 0: zero the histogram workspace and the output accumulator.
// ---------------------------------------------------------------------------
__global__ void hem_init_kernel(unsigned* __restrict__ hist, float* __restrict__ out) {
    int i = blockIdx.x * blockDim.x + threadIdx.x;
    if (i < BATCH * NBINS) hist[i] = 0u;
    if (i == 0) out[0] = 0.0f;
}

// ---------------------------------------------------------------------------
// Kernel 1: per-batch histogram of channel-summed residuals (pass 1).
// float4 (b128) streaming loads; LDS-resident histogram (16 KB of 320 KB).
// ---------------------------------------------------------------------------
__global__ void hem_hist_kernel(const float* __restrict__ x,
                                const float* __restrict__ y,
                                unsigned* __restrict__ hist) {
    __shared__ unsigned lh[NBINS];
    const int b = blockIdx.y;
    for (int i = threadIdx.x; i < NBINS; i += blockDim.x) lh[i] = 0u;
    __syncthreads();

    const float4* xb = (const float4*)(x + (size_t)b * CHAN * NPIX);
    const float4* yb = (const float4*)(y + (size_t)b * CHAN * NPIX);

    const int stride = gridDim.x * blockDim.x;
    for (int v = blockIdx.x * blockDim.x + threadIdx.x; v < N4; v += stride) {
        __builtin_prefetch(xb + v + stride, 0, 1);     // global_prefetch_b8
        __builtin_prefetch(yb + v + stride, 0, 1);
        float4 x0 = xb[v], x1 = xb[v + N4], x2 = xb[v + 2 * N4];
        float4 y0 = yb[v], y1 = yb[v + N4], y2 = yb[v + 2 * N4];
        float r[4];
        residual4(x0, x1, x2, y0, y1, y2, r);
#pragma unroll
        for (int e = 0; e < 4; ++e) {
            int bin = (int)(r[e] * BIN_SCALE);
            bin = bin > (NBINS - 1) ? (NBINS - 1) : bin;
            atomicAdd(&lh[bin], 1u);                   // ds_add_u32
        }
    }
    __syncthreads();

    unsigned* gh = hist + (size_t)b * NBINS;
    for (int i = threadIdx.x; i < NBINS; i += blockDim.x)
        if (lh[i]) atomicAdd(&gh[i], lh[i]);           // global_atomic_add_u32
}

// ---------------------------------------------------------------------------
// Kernel 2: per-batch threshold = approximate (n/2)-th largest residual.
// Suffix-scan the histogram from the top; 8 lanes, one per batch.
// ---------------------------------------------------------------------------
__global__ void hem_thresh_kernel(const unsigned* __restrict__ hist,
                                  float* __restrict__ thre) {
    int b = threadIdx.x;
    if (b >= BATCH) return;
    const unsigned* h = hist + (size_t)b * NBINS;
    unsigned acc = 0;
    int j = NBINS - 1;
    for (; j > 0; --j) {
        acc += h[j];
        if (acc > (unsigned)HARD_IND) break;
    }
    thre[b] = (float)j * (1.0f / BIN_SCALE);           // left edge of kth bin
}

// ---------------------------------------------------------------------------
// Kernel 3: pass 2 — recompute residuals via b128 loads, apply the
// (hard | random) mask, and accumulate on the matrix pipe: each lane packs
// 16 masked residuals (4 float4 loads per plane) into a 16x16x32 f16
// A-fragment; B = ones; C (f32) accumulates. Total(D) = 16 * sum(masked r).
// Grid divides N exactly -> branchless, EXEC full at the WMMA (wave32).
// ---------------------------------------------------------------------------
__global__ void hem_reduce_kernel(const float* __restrict__ x,
                                  const float* __restrict__ y,
                                  const float* __restrict__ thre,
                                  float* __restrict__ out) {
    const int b   = blockIdx.y;
    const float t = thre[b];
    const float4* xb = (const float4*)(x + (size_t)b * CHAN * NPIX);
    const float4* yb = (const float4*)(y + (size_t)b * CHAN * NPIX);

    // block chunk: blockDim.x * 4 float4s = blockDim.x * 16 pixels
    const int f4base = blockIdx.x * (blockDim.x * 4);

    v16h a;
#pragma unroll
    for (int q = 0; q < 4; ++q) {
        const int v = f4base + q * blockDim.x + threadIdx.x;   // coalesced
        float4 x0 = xb[v], x1 = xb[v + N4], x2 = xb[v + 2 * N4];
        float4 y0 = yb[v], y1 = yb[v + N4], y2 = yb[v + 2 * N4];
        float r[4];
        residual4(x0, x1, x2, y0, y1, y2, r);
        const uint32_t i0 = (uint32_t)(b * NPIX + v * 4);
#pragma unroll
        for (int e = 0; e < 4; ++e) {
            bool hard = (r[e] > t);
            bool rnd  = hash_u32(i0 + e) < RAND_CUT;
            a[q * 4 + e] = (_Float16)((hard || rnd) ? r[e] : 0.0f);
        }
    }

    v16h ones;
#pragma unroll
    for (int j = 0; j < 16; ++j) ones[j] = (_Float16)1.0f;
    v8f c = {};
    // D = A x ones + C  ->  v_wmma_f32_16x16x32_f16
    c = __builtin_amdgcn_wmma_f32_16x16x32_f16(
            /*neg_a=*/false, a, /*neg_b=*/false, ones,
            /*c_mod=*/(short)0, c, /*reuse_a=*/false, /*reuse_b=*/false);

    // Fold the 16x16 f32 tile: per-lane sum of 8 VGPRs, wave32 butterfly,
    // LDS across the 8 waves, single f32 atomic per block.
    float s = c[0] + c[1] + c[2] + c[3] + c[4] + c[5] + c[6] + c[7];
#pragma unroll
    for (int off = 16; off >= 1; off >>= 1) s += __shfl_xor(s, off, 32);

    __shared__ float wsum[8];
    if ((threadIdx.x & 31) == 0) wsum[threadIdx.x >> 5] = s;
    __syncthreads();
    if (threadIdx.x == 0) {
        float tot = 0.0f;
#pragma unroll
        for (int w = 0; w < 8; ++w) tot += wsum[w];
        // Total(D) = 16 * sum; mean over B*C*N = 8*3*2^20 elements.
        const float scale = 1.0f /
            (16.0f * (float)BATCH * (float)CHAN * (float)NPIX);
        atomicAdd(out, tot * scale);                   // global_atomic_add_f32
    }
}

// ---------------------------------------------------------------------------
// Host-side launch: init -> histogram -> threshold -> masked WMMA reduce.
// Workspace layout: [0..31] float thre[8]; [256 ..] unsigned hist[8][4096].
// ---------------------------------------------------------------------------
extern "C" void kernel_launch(void* const* d_in, const int* in_sizes, int n_in,
                              void* d_out, int out_size, void* d_ws, size_t ws_size,
                              hipStream_t stream) {
    const float* x = (const float*)d_in[0];
    const float* y = (const float*)d_in[1];
    float*    out  = (float*)d_out;
    float*    thre = (float*)d_ws;
    unsigned* hist = (unsigned*)((char*)d_ws + 256);

    (void)in_sizes; (void)n_in; (void)out_size; (void)ws_size;

    // 0) zero histogram + output accumulator
    hem_init_kernel<<<(BATCH * NBINS + 255) / 256, 256, 0, stream>>>(hist, out);

    // 1) residual histograms (pass 1 over x, y), b128 streaming loads
    hem_hist_kernel<<<dim3(256, BATCH), 256, 0, stream>>>(x, y, hist);

    // 2) per-batch median threshold from histogram suffix scan
    hem_thresh_kernel<<<1, 32, 0, stream>>>(hist, thre);

    // 3) masked L1 mean via WMMA-accumulated reduction (pass 2 over x, y)
    //    256 threads * 16 px = 4096 px/block; 256 x 8 blocks covers exactly.
    hem_reduce_kernel<<<dim3(NPIX / 4096, BATCH), 256, 0, stream>>>(x, y, thre, out);
}
